// StyleBlock_72499047957302
// MI455X (gfx1250) — compile-verified, matrix operand order checked by reading, not run on hardware
//
#include <hip/hip_runtime.h>

typedef __attribute__((ext_vector_type(16))) _Float16 v16h;
typedef __attribute__((ext_vector_type(8)))  float    v8f;

#define LRELU 0.2f
#define NB   4
#define COUT 256
#define WD   512
#define HOUT 128
#define NPIX (HOUT*HOUT)   // 16384 per batch image
#define PSTR 40            // LDS halves per halo pixel (32 data + 8 pad, bank-tiled)

// ---------------------------------------------------------------------------
// 1) styles: s1[4,512], s2[4,256], s3[4,256] packed into one buffer
// ---------------------------------------------------------------------------
__global__ void styles_kernel(const float* __restrict__ w,
                              const float* __restrict__ a1w, const float* __restrict__ a1b,
                              const float* __restrict__ a2w, const float* __restrict__ a2b,
                              const float* __restrict__ a3w, const float* __restrict__ a3b,
                              float* __restrict__ styles) {
  int gid = blockIdx.x * blockDim.x + threadIdx.x;
  if (gid >= 4096) return;
  const float *aw, *ab; int b, c;
  if (gid < 2048)      { b = gid >> 9;             c = gid & 511; aw = a1w; ab = a1b; }
  else if (gid < 3072) { int i = gid - 2048; b = i >> 8; c = i & 255; aw = a2w; ab = a2b; }
  else                 { int i = gid - 3072; b = i >> 8; c = i & 255; aw = a3w; ab = a3b; }
  float acc = ab[c];
  const float* wr = w + b * WD;
  const float* ar = aw + c * WD;
  for (int d = 0; d < WD; ++d) acc += wr[d] * ar[d];
  styles[gid] = acc;
}

// ---------------------------------------------------------------------------
// 2) demod[b,co] = rsqrt( sum_{ci,tap} (W[co,ci,tap]*s[b,ci])^2 + 1e-8 )
// ---------------------------------------------------------------------------
__global__ void demod_kernel(const float* __restrict__ w1,
                             const float* __restrict__ w2,
                             const float* __restrict__ styles,
                             float* __restrict__ demod) {
  __shared__ float red[256];
  int r = blockIdx.x, t = threadIdx.x;
  const float *wgt, *s; int Cin;
  if (r < 1024) { int co = r & 255; int b = r >> 8;
                  wgt = w1 + (size_t)co * 512 * 9; s = styles + b * 512;        Cin = 512; }
  else          { int rr = r - 1024; int co = rr & 255; int b = rr >> 8;
                  wgt = w2 + (size_t)co * 256 * 9; s = styles + 2048 + b * 256; Cin = 256; }
  float p = 0.f;
  int n = Cin * 9;
  for (int i = t; i < n; i += 256) { float v = wgt[i] * s[i / 9]; p += v * v; }
  red[t] = p; __syncthreads();
  for (int st = 128; st > 0; st >>= 1) { if (t < st) red[t] += red[t + st]; __syncthreads(); }
  if (t == 0) demod[r] = rsqrtf(red[0] + 1e-8f);
}

// ---------------------------------------------------------------------------
// 3) permute conv weights [Cout][Cin][3][3] f32 -> [tap][Cout][Cin] f16
// ---------------------------------------------------------------------------
__global__ void wcvt_kernel(const float* __restrict__ w1, const float* __restrict__ w2,
                            _Float16* __restrict__ W1h, _Float16* __restrict__ W2h) {
  int gid = blockIdx.x * blockDim.x + threadIdx.x;
  const int n1 = 256 * 512 * 9, n2 = 256 * 256 * 9;
  if (gid < n1) {
    int co = gid / (512 * 9); int rem = gid % (512 * 9); int ci = rem / 9; int tap = rem % 9;
    W1h[((size_t)tap * 256 + co) * 512 + ci] = (_Float16)w1[gid];
  } else if (gid < n1 + n2) {
    int g = gid - n1;
    int co = g / (256 * 9); int rem = g % (256 * 9); int ci = rem / 9; int tap = rem % 9;
    W2h[((size_t)tap * 256 + co) * 256 + ci] = (_Float16)w2[g];
  }
}

// ---------------------------------------------------------------------------
// 4) fused half-pixel bilinear x2 upsample + style modulate + f16, NHWC out
// ---------------------------------------------------------------------------
__global__ void upsample_mod_kernel(const float* __restrict__ x,
                                    const float* __restrict__ styles,
                                    _Float16* __restrict__ Xmod) {
  int pid = blockIdx.x;                 // b*16384 + y*128 + x
  int t   = threadIdx.x;
  int b = pid >> 14; int rem = pid & 16383; int y = rem >> 7; int xo = rem & 127;
  int iy0 = (y - 1) >> 1;  float fy = (y & 1) ? 0.25f : 0.75f;
  int ix0 = (xo - 1) >> 1; float fx = (xo & 1) ? 0.25f : 0.75f;
  int iy1 = iy0 + 1; if (iy0 < 0) iy0 = 0; if (iy1 > 63) iy1 = 63;
  int ix1 = ix0 + 1; if (ix0 < 0) ix0 = 0; if (ix1 > 63) ix1 = 63;
  const float* s = styles + b * 512;
  _Float16* outp = Xmod + (size_t)pid * 512;
  for (int ci = t; ci < 512; ci += 256) {
    const float* xp = x + ((size_t)(b * 512 + ci)) * 64 * 64;
    float v00 = xp[iy0 * 64 + ix0], v01 = xp[iy0 * 64 + ix1];
    float v10 = xp[iy1 * 64 + ix0], v11 = xp[iy1 * 64 + ix1];
    float v0 = v00 * (1.f - fx) + v01 * fx;
    float v1 = v10 * (1.f - fx) + v11 * fx;
    float v  = v0  * (1.f - fy) + v1  * fy;
    outp[ci] = (_Float16)(v * s[ci]);
  }
}

// ---------------------------------------------------------------------------
// 5/6) 3x3 modulated conv: halo-tiled implicit GEMM, all 9 taps per LDS stage.
//  Block: 256 thr (8 waves). Block tile: all 256 Cout x 64 pixels of one row.
//  Wave w: Cout rows [32w, 32w+32) -> 2 A fragments, 8 v8f accumulators.
//  LDS: double-buffered 3x66-pixel halo tile, PSTR-padded for bank tiling.
//  Per K-chunk: 1 barrier, 9 taps x 8 WMMA = 72 WMMA per wave.
// ---------------------------------------------------------------------------
__global__ __launch_bounds__(256) void wmma_conv3x3(
    const _Float16* __restrict__ Xin,   // NHWC [B,128,128,Cin], style folded
    const _Float16* __restrict__ Wt,    // [9][COUT][Cin] f16
    const float* __restrict__ demod,    // [B][COUT]
    const float* __restrict__ bias,     // [COUT]
    const float* __restrict__ snext,    // [B][COUT] next-layer style
    _Float16* __restrict__ Ynext,       // NHWC [B,128,128,COUT]
    float* __restrict__ Yf32,           // NCHW final h (or nullptr)
    int Cin) {
  const int TPX = 3 * 66;               // 3 rows x (64+2) halo pixels
  __shared__ _Float16 Bs[2][TPX * PSTR];

  int nt = blockIdx.x;                  // 0..1023: (b, y, x-half)
  int b = nt >> 8; int rem = nt & 255; int y = rem >> 1; int x0 = (rem & 1) << 6;
  int t = threadIdx.x;
  int wave = t >> 5, lane = t & 31;
  int lhalf = (lane & 16) ? 8 : 0;      // ISA 16-bit A/B K interleave
  int nl = lane & 15;
  int m0 = wave * 32 + nl;              // first A row of this lane

  v8f acc[2][4];
  v8f zero = {0.f, 0.f, 0.f, 0.f, 0.f, 0.f, 0.f, 0.f};
#pragma unroll
  for (int mf = 0; mf < 2; ++mf)
#pragma unroll
    for (int ns = 0; ns < 4; ++ns) acc[mf][ns] = zero;

  int kchunks = Cin >> 5;

  // B halo stage: 198 px * 4 kg-groups = 792 uint4 loads per chunk.
  uint4 breg[4];
  auto fetchB = [&](int kc) {
#pragma unroll
    for (int u = 0; u < 4; ++u) {
      int idx = t + u * 256;
      if (idx < TPX * 4) {
        int p = idx >> 2, kg = (idx & 3) << 3;
        int r = p / 66, col = p - r * 66;
        int ys = y - 1 + r, xs = x0 - 1 + col;
        if (ys >= 0 && ys < 128 && xs >= 0 && xs < 128) {
          breg[u] = *(const uint4*)(Xin +
              ((size_t)((b * 128 + ys) * 128 + xs)) * Cin + kc * 32 + kg);
        } else { breg[u].x = breg[u].y = breg[u].z = breg[u].w = 0u; }
      }
    }
  };
  auto storeB = [&](int buf) {
#pragma unroll
    for (int u = 0; u < 4; ++u) {
      int idx = t + u * 256;
      if (idx < TPX * 4) {
        int p = idx >> 2, kg = (idx & 3) << 3;
        *(uint4*)&Bs[buf][p * PSTR + kg] = breg[u];
      }
    }
  };

  fetchB(0);
  storeB(0);

  for (int kc = 0; kc < kchunks; ++kc) {
    bool more = (kc + 1 < kchunks);
    if (more) fetchB(kc + 1);           // overlap next global fetch with compute
    __syncthreads();                    // buf[kc&1] visible; prev readers done
    const _Float16* bb = Bs[kc & 1];
    if (more) __builtin_prefetch(Wt + (size_t)m0 * Cin + (kc + 1) * 32, 0, 0);

#pragma unroll
    for (int tap = 0; tap < 9; ++tap) {
      int dyr = (tap / 3) * 66;         // halo row offset (dy+1)
      int dx  = tap % 3;                // halo col offset (dx+1)
      union F { uint4 u[2]; v16h h; } a0, a1;
      const _Float16* ap = Wt + ((size_t)(tap * COUT + m0)) * Cin + kc * 32 + lhalf;
      a0.u[0] = *(const uint4*)(ap);
      a0.u[1] = *(const uint4*)(ap + 16);
      const _Float16* ap1 = ap + (size_t)16 * Cin;
      a1.u[0] = *(const uint4*)(ap1);
      a1.u[1] = *(const uint4*)(ap1 + 16);
#pragma unroll
      for (int hhalf = 0; hhalf < 2; ++hhalf) {
        int nsb = hhalf * 2;
        union F bf0, bf1;
        const _Float16* bp0 = bb + (dyr + nsb * 16 + nl + dx) * PSTR + lhalf;
        const _Float16* bp1 = bp0 + 16 * PSTR;
        bf0.u[0] = *(const uint4*)(bp0);
        bf0.u[1] = *(const uint4*)(bp0 + 16);
        bf1.u[0] = *(const uint4*)(bp1);
        bf1.u[1] = *(const uint4*)(bp1 + 16);
        acc[0][nsb]     = __builtin_amdgcn_wmma_f32_16x16x32_f16(
            false, a0.h, false, bf0.h, (short)0, acc[0][nsb],     false, false);
        acc[1][nsb]     = __builtin_amdgcn_wmma_f32_16x16x32_f16(
            false, a1.h, false, bf0.h, (short)0, acc[1][nsb],     false, false);
        acc[0][nsb + 1] = __builtin_amdgcn_wmma_f32_16x16x32_f16(
            false, a0.h, false, bf1.h, (short)0, acc[0][nsb + 1], false, false);
        acc[1][nsb + 1] = __builtin_amdgcn_wmma_f32_16x16x32_f16(
            false, a1.h, false, bf1.h, (short)0, acc[1][nsb + 1], false, false);
      }
    }
    if (more) storeB((kc + 1) & 1);     // safe: other buffer, post-barrier
  }

  // epilogue: vgpr j holds M=j (lanes 0-15) / M=j+8 (lanes 16-31).
  // each v8f accumulator = 8 consecutive co at one pixel -> pack 16B store.
  int mhi = (lane & 16) ? 8 : 0;
#pragma unroll
  for (int mf = 0; mf < 2; ++mf) {
    int cobase = wave * 32 + mf * 16 + mhi;
#pragma unroll
    for (int ns = 0; ns < 4; ++ns) {
      int px = x0 + ns * 16 + nl;
      size_t pix = (size_t)((b * 128 + y) * 128 + px);
      union { uint4 u; _Float16 h[8]; } pk;
#pragma unroll
      for (int j = 0; j < 8; ++j) {
        int co = cobase + j;
        float v = acc[mf][ns][j] * demod[b * COUT + co] + bias[co];
        v = (v > 0.f) ? v : LRELU * v;
        if (Yf32) Yf32[((size_t)(b * COUT + co) * 128 + y) * 128 + px] = v;
        pk.h[j] = (_Float16)(v * snext[b * COUT + co]);
      }
      *(uint4*)&Ynext[pix * COUT + cobase] = pk.u;
    }
  }
}

// ---------------------------------------------------------------------------
// 7) to-RGB 1x1 (s3 already folded into Y2mod)
// ---------------------------------------------------------------------------
__global__ void rgb_kernel(const _Float16* __restrict__ Y2mod,
                           const float* __restrict__ w3,
                           const float* __restrict__ b3,
                           float* __restrict__ rgb_out) {
  int pid = blockIdx.x * blockDim.x + threadIdx.x;
  if (pid >= NB * NPIX) return;
  int b = pid >> 14; int rem = pid & 16383;
  const uint4* yp4 = (const uint4*)(Y2mod + (size_t)pid * COUT);
  float a0 = b3[0], a1 = b3[1], a2 = b3[2];
  for (int it = 0; it < COUT / 8; ++it) {
    union { uint4 u; _Float16 h[8]; } blk;
    blk.u = yp4[it];
#pragma unroll
    for (int j = 0; j < 8; ++j) {
      float v = (float)blk.h[j];
      int co = it * 8 + j;
      a0 += w3[0 * COUT + co] * v;
      a1 += w3[1 * COUT + co] * v;
      a2 += w3[2 * COUT + co] * v;
    }
  }
  a0 = (a0 > 0.f) ? a0 : LRELU * a0;
  a1 = (a1 > 0.f) ? a1 : LRELU * a1;
  a2 = (a2 > 0.f) ? a2 : LRELU * a2;
  rgb_out[((size_t)(b * 3 + 0) * NPIX) + rem] = a0;
  rgb_out[((size_t)(b * 3 + 1) * NPIX) + rem] = a1;
  rgb_out[((size_t)(b * 3 + 2) * NPIX) + rem] = a2;
}

// ---------------------------------------------------------------------------
extern "C" void kernel_launch(void* const* d_in, const int* in_sizes, int n_in,
                              void* d_out, int out_size, void* d_ws, size_t ws_size,
                              hipStream_t stream) {
  const float* x   = (const float*)d_in[0];
  const float* w   = (const float*)d_in[1];
  const float* w1  = (const float*)d_in[2];
  const float* b1  = (const float*)d_in[3];
  const float* a1w = (const float*)d_in[4];
  const float* a1b = (const float*)d_in[5];
  const float* w2  = (const float*)d_in[6];
  const float* b2  = (const float*)d_in[7];
  const float* a2w = (const float*)d_in[8];
  const float* a2b = (const float*)d_in[9];
  const float* w3  = (const float*)d_in[10];
  const float* b3  = (const float*)d_in[11];
  const float* a3w = (const float*)d_in[12];
  const float* a3b = (const float*)d_in[13];

  char* ws = (char*)d_ws;
  float*    styles = (float*)(ws + 0);                 // 4096 f32: s1|s2|s3
  float*    demod  = (float*)(ws + 16384);             // 2048 f32: demod1|demod2
  _Float16* W1h    = (_Float16*)(ws + 24576);          // 9*256*512 f16
  _Float16* W2h    = (_Float16*)(ws + 2383872);        // 9*256*256 f16
  _Float16* Xmod   = (_Float16*)(ws + 3563520);        // [4,128,128,512] f16
  _Float16* Y1mod  = (_Float16*)(ws + 70672384);       // [4,128,128,256] f16
  _Float16* Y2mod  = (_Float16*)(ws + 104226816);      // [4,128,128,256] f16

  float* out_h   = (float*)d_out;                      // [4,256,128,128]
  float* out_rgb = out_h + (size_t)NB * COUT * NPIX;   // [4,3,128,128]

  styles_kernel<<<16, 256, 0, stream>>>(w, a1w, a1b, a2w, a2b, a3w, a3b, styles);
  demod_kernel<<<2048, 256, 0, stream>>>(w1, w2, styles, demod);
  wcvt_kernel<<<(256 * 512 * 9 + 256 * 256 * 9 + 255) / 256, 256, 0, stream>>>(w1, w2, W1h, W2h);
  upsample_mod_kernel<<<NB * NPIX, 256, 0, stream>>>(x, styles, Xmod);

  // conv1: Cin=512; fold s2 into output for conv2
  wmma_conv3x3<<<dim3(1024), 256, 0, stream>>>(
      Xmod, W1h, demod, b1, styles + 2048, Y1mod, nullptr, 512);
  // conv2: Cin=256; writes final f32 h, folds s3 into Y2mod
  wmma_conv3x3<<<dim3(1024), 256, 0, stream>>>(
      Y1mod, W2h, demod + 1024, b2, styles + 3072, Y2mod, out_h, 256);

  rgb_kernel<<<(NB * NPIX + 255) / 256, 256, 0, stream>>>(Y2mod, w3, b3, out_rgb);
}
